// Block_11132555231612
// MI455X (gfx1250) — matrix operand order repaired and checked by reading, compiled, not run on hardware
//
#include <hip/hip_runtime.h>
#include <hip/hip_bf16.h>
#include <math.h>

// ---------------------------------------------------------------------------
// ViT transformer block forward on gfx1250 (MI455X).
// All matmuls via v_wmma_f32_16x16x32_bf16 (f32 accumulate).
// GEMMs: per-wave 32x64 tiles, fully-unrolled register double-buffered loop.
// Attention: flash-style, online softmax, async-to-LDS K staging.
// ---------------------------------------------------------------------------

typedef __bf16 bf16_t;
typedef __attribute__((ext_vector_type(16))) __bf16 v16bf;
typedef __attribute__((ext_vector_type(8)))  __bf16 v8bf;
typedef __attribute__((ext_vector_type(8)))  float  v8f;
typedef __attribute__((ext_vector_type(4)))  int    v4i;

#define DIMC   768
#define HIDDEN 3072
#define NSEQ   1024
#define NBATCH 16
#define NHEAD  12
#define HD     64
#define MROWS  (NBATCH * NSEQ)   /* 16384 */
#define BHTOT  (NBATCH * NHEAD)  /* 192   */

#if defined(__gfx1250__) && __has_builtin(__builtin_amdgcn_global_load_async_to_lds_b128)
#define USE_ASYNC_KSTAGE 1
#else
#define USE_ASYNC_KSTAGE 0
#endif

union ABu { v16bf v; v8bf h[2]; };

// A-fragment (16x32 bf16): lane holds row m = lane%16;
// elements 0..15 = K[k + 16*half .. +15]
static __device__ inline v16bf load_a_frag(const bf16_t* __restrict__ row, int kbase, int half) {
  ABu u;
  u.h[0] = *(const v8bf*)(row + kbase + half * 16);
  u.h[1] = *(const v8bf*)(row + kbase + half * 16 + 8);
  return u.v;
}

static __device__ inline v8f wmma_bf16(v16bf a, v16bf b, v8f c) {
  return __builtin_amdgcn_wmma_f32_16x16x32_bf16(false, a, false, b, (short)0, c, false, false);
}

#if USE_ASYNC_KSTAGE
typedef __attribute__((address_space(1))) char* gas1_t;
typedef __attribute__((address_space(3))) char* las3_t;
typedef __attribute__((address_space(1))) v4i*  g4_t;
typedef __attribute__((address_space(3))) v4i*  l4_t;
// Copy one contiguous 4KB tile (32 key rows x 128B) global -> LDS, async.
// Per issue: 32 lanes x 16B = 512B; global and LDS offsets coincide.
static __device__ inline void async_copy_4k(const bf16_t* g, bf16_t* l, int lane) {
  gas1_t gp = (gas1_t)(char*)g;
  las3_t lp = (las3_t)(char*)l;
#pragma unroll
  for (int i = 0; i < 8; ++i) {
    __builtin_amdgcn_global_load_async_to_lds_b128((g4_t)(gp + i * 512 + lane * 16),
                                                   (l4_t)(lp + i * 512 + lane * 16), 0, 0);
  }
}

static __device__ inline void wait_async_le8() {
#if __has_builtin(__builtin_amdgcn_s_wait_asynccnt)
  asm volatile("" ::: "memory");
  __builtin_amdgcn_s_wait_asynccnt(8);
  asm volatile("" ::: "memory");
#else
  asm volatile("s_wait_asynccnt 0x8" ::: "memory");
#endif
}
#endif

// ---------------------------------------------------------------------------
// f32 -> bf16 conversion (weights)
// ---------------------------------------------------------------------------
__global__ void f32_to_bf16_kernel(const float* __restrict__ src, bf16_t* __restrict__ dst, int n) {
  int i = blockIdx.x * 256 + threadIdx.x;
  if (i < n) dst[i] = (bf16_t)src[i];
}

// ---------------------------------------------------------------------------
// LayerNorm over rows of 768, output bf16. One block (256 threads) per row.
// ---------------------------------------------------------------------------
__global__ __launch_bounds__(256) void ln_bf16_kernel(const float* __restrict__ x,
                                                      const float* __restrict__ g,
                                                      const float* __restrict__ b,
                                                      bf16_t* __restrict__ out) {
  int row = blockIdx.x;
  int t = threadIdx.x;
  const float* xr = x + (size_t)row * DIMC;
  float v0 = xr[t], v1 = xr[t + 256], v2 = xr[t + 512];
  float s  = v0 + v1 + v2;
  float s2 = v0 * v0 + v1 * v1 + v2 * v2;
#pragma unroll
  for (int m = 1; m <= 16; m <<= 1) {
    s  += __shfl_xor(s,  m, 32);
    s2 += __shfl_xor(s2, m, 32);
  }
  __shared__ float red0[8], red1[8];
  int wid = t >> 5, lane = t & 31;
  if (lane == 0) { red0[wid] = s; red1[wid] = s2; }
  __syncthreads();
  float ts = 0.f, ts2 = 0.f;
#pragma unroll
  for (int i = 0; i < 8; ++i) { ts += red0[i]; ts2 += red1[i]; }
  float mu  = ts * (1.0f / DIMC);
  float var = ts2 * (1.0f / DIMC) - mu * mu;
  float inv = rsqrtf(var + 1e-5f);
  bf16_t* orow = out + (size_t)row * DIMC;
  orow[t]       = (bf16_t)((v0 - mu) * inv * g[t]       + b[t]);
  orow[t + 256] = (bf16_t)((v1 - mu) * inv * g[t + 256] + b[t + 256]);
  orow[t + 512] = (bf16_t)((v2 - mu) * inv * g[t + 512] + b[t + 512]);
}

// ---------------------------------------------------------------------------
// WMMA GEMM tile: one wave computes D[32 x 64] = A[32 x K] * W^T tile.
// A: [M,K] bf16 row-major. W: [N,K] bf16 row-major (torch Linear weight).
// K is a compile-time constant; the k-loop is fully unrolled so the
// double-buffer rotation is pure SSA renaming (no acc copies, no hazard nops)
// while next-step fragments are loaded ahead of the current 8 WMMAs.
// acc[i*4+t] covers rows m0+i*16..+15, cols n0+t*16..+15.
// ---------------------------------------------------------------------------
template <int K>
static __device__ inline void gemm_tile_32x64(const bf16_t* __restrict__ A,
                                              const bf16_t* __restrict__ W,
                                              int m0, int n0,
                                              int half, int lr, v8f acc[8]) {
  const bf16_t* arow0 = A + (size_t)(m0 + lr) * K;
  const bf16_t* arow1 = A + (size_t)(m0 + 16 + lr) * K;
  const bf16_t* w0 = W + (size_t)(n0 + 0 * 16 + lr) * K + half * 16;
  const bf16_t* w1 = W + (size_t)(n0 + 1 * 16 + lr) * K + half * 16;
  const bf16_t* w2 = W + (size_t)(n0 + 2 * 16 + lr) * K + half * 16;
  const bf16_t* w3 = W + (size_t)(n0 + 3 * 16 + lr) * K + half * 16;

  v16bf a0 = load_a_frag(arow0, 0, half);
  v16bf a1 = load_a_frag(arow1, 0, half);
  v16bf b0 = *(const v16bf*)(w0);
  v16bf b1 = *(const v16bf*)(w1);
  v16bf b2 = *(const v16bf*)(w2);
  v16bf b3 = *(const v16bf*)(w3);

#pragma unroll
  for (int k = 32; k < K; k += 32) {
    v16bf na0 = load_a_frag(arow0, k, half);
    v16bf na1 = load_a_frag(arow1, k, half);
    v16bf nb0 = *(const v16bf*)(w0 + k);
    v16bf nb1 = *(const v16bf*)(w1 + k);
    v16bf nb2 = *(const v16bf*)(w2 + k);
    v16bf nb3 = *(const v16bf*)(w3 + k);
    acc[0] = wmma_bf16(a0, b0, acc[0]);
    acc[1] = wmma_bf16(a0, b1, acc[1]);
    acc[2] = wmma_bf16(a0, b2, acc[2]);
    acc[3] = wmma_bf16(a0, b3, acc[3]);
    acc[4] = wmma_bf16(a1, b0, acc[4]);
    acc[5] = wmma_bf16(a1, b1, acc[5]);
    acc[6] = wmma_bf16(a1, b2, acc[6]);
    acc[7] = wmma_bf16(a1, b3, acc[7]);
    a0 = na0; a1 = na1; b0 = nb0; b1 = nb1; b2 = nb2; b3 = nb3;
  }
  acc[0] = wmma_bf16(a0, b0, acc[0]);
  acc[1] = wmma_bf16(a0, b1, acc[1]);
  acc[2] = wmma_bf16(a0, b2, acc[2]);
  acc[3] = wmma_bf16(a0, b3, acc[3]);
  acc[4] = wmma_bf16(a1, b0, acc[4]);
  acc[5] = wmma_bf16(a1, b1, acc[5]);
  acc[6] = wmma_bf16(a1, b2, acc[6]);
  acc[7] = wmma_bf16(a1, b3, acc[7]);
}

static __device__ inline void zero_acc(v8f acc[8]) {
  v8f z = {};
#pragma unroll
  for (int i = 0; i < 8; ++i) acc[i] = z;
}

// ---------------------------------------------------------------------------
// QKV GEMM: [16384 x 768] x [2304 x 768]^T, scatter epilogue:
//   q,k -> [bh, n, d] bf16 ; v -> transposed [bh, d, n] bf16 (for PV B-frags)
// ---------------------------------------------------------------------------
__global__ __launch_bounds__(256) void gemm_qkv_kernel(const bf16_t* __restrict__ A,
                                                       const bf16_t* __restrict__ W,
                                                       bf16_t* __restrict__ qo,
                                                       bf16_t* __restrict__ ko,
                                                       bf16_t* __restrict__ vto) {
  const int NT = (3 * DIMC) / 64;  // 36 column tiles
  int wid = threadIdx.x >> 5, lane = threadIdx.x & 31;
  int half = lane >> 4, lr = lane & 15;
  long gw = (long)blockIdx.x * 8 + wid;
  int m0 = (int)(gw / NT) * 32;
  int n0 = (int)(gw % NT) * 64;
  v8f acc[8];
  zero_acc(acc);
  gemm_tile_32x64<DIMC>(A, W, m0, n0, half, lr, acc);
#pragma unroll
  for (int i = 0; i < 2; ++i) {
#pragma unroll
    for (int t = 0; t < 4; ++t) {
#pragma unroll
      for (int j = 0; j < 8; ++j) {
        int m   = m0 + i * 16 + j + 8 * half;
        int col = n0 + t * 16 + lr;
        int which = col / DIMC;
        int c = col % DIMC;
        int hh = c >> 6, d = c & 63;
        int bb = m >> 10, nn = m & 1023;
        int bh = bb * NHEAD + hh;
        bf16_t bv = (bf16_t)acc[i * 4 + t][j];
        if (which == 0)      qo[((size_t)bh * NSEQ + nn) * HD + d] = bv;
        else if (which == 1) ko[((size_t)bh * NSEQ + nn) * HD + d] = bv;
        else                 vto[((size_t)bh * HD + d) * NSEQ + nn] = bv;
      }
    }
  }
}

// ---------------------------------------------------------------------------
// Flash attention: one wave handles 16 query rows for one (b,h).
// Per 32-key step: S(16x32) via 4 WMMAs, online softmax (shfl_xor within
// 16-lane halves matches C/D layout), P staged D-layout->A-layout via LDS,
// O(16x64) += P*V via 4 WMMAs (V pre-transposed).
// Next K-tile (contiguous 4KB) is staged into LDS with
// global_load_async_to_lds_b128, double-buffered, drained via s_wait_asynccnt.
// ---------------------------------------------------------------------------
__global__ __launch_bounds__(256) void attn_kernel(const bf16_t* __restrict__ Q,
                                                   const bf16_t* __restrict__ Km,
                                                   const bf16_t* __restrict__ Vt,
                                                   bf16_t* __restrict__ O) {
  __shared__ __align__(32) bf16_t Pbuf[8][16 * 32];
#if USE_ASYNC_KSTAGE
  __shared__ __align__(32) bf16_t Kstage[8][2][32 * HD];  // 2 x 4KB per wave
#endif
  int wid = threadIdx.x >> 5, lane = threadIdx.x & 31;
  int half = lane >> 4, lr = lane & 15;
  long gw = (long)blockIdx.x * 8 + wid;
  int bh = (int)(gw >> 6);       // 64 query tiles per (b,h)
  int qt = (int)(gw & 63);

  const bf16_t* qrow = Q + ((size_t)bh * NSEQ + qt * 16 + lr) * HD;
  v16bf aq0 = load_a_frag(qrow, 0, half);
  v16bf aq1 = load_a_frag(qrow, 32, half);

  v8f o0 = {}, o1 = {}, o2 = {}, o3 = {};
  float mrun[8], lrun[8];
#pragma unroll
  for (int j = 0; j < 8; ++j) { mrun[j] = -1e30f; lrun[j] = 0.0f; }

  bf16_t* pb = &Pbuf[wid][0];
  const bf16_t* kbase = Km + (size_t)bh * NSEQ * HD;
  const float scale = 0.125f;  // hd=64 -> 1/sqrt(64)

#if USE_ASYNC_KSTAGE
  bf16_t* kst = &Kstage[wid][0][0];
  async_copy_4k(kbase, kst, lane);          // stage tile 0 into buffer 0
#endif

#pragma unroll 2
  for (int kt = 0; kt < NSEQ / 32; ++kt) {
    // ---- obtain K tile (32 keys x 64) ----
#if USE_ASYNC_KSTAGE
    int ktn = (kt + 1) & (NSEQ / 32 - 1);   // wraps on last iter (kept uniform)
    async_copy_4k(kbase + (size_t)ktn * 32 * HD, kst + ((kt + 1) & 1) * (32 * HD), lane);
    wait_async_le8();                        // tile kt has landed (in-order)
    const bf16_t* kl = kst + (kt & 1) * (32 * HD);
#else
    const bf16_t* kl = kbase + (size_t)kt * 32 * HD;
    __builtin_prefetch(kl + 32 * HD, 0, 3);
#endif

    // ---- S = Q * K^T over 32 keys, two 16-key column tiles ----
    v8f s0 = {}, s1 = {};
    s0 = wmma_bf16(aq0, *(const v16bf*)(kl + (size_t)lr * HD + half * 16), s0);
    s0 = wmma_bf16(aq1, *(const v16bf*)(kl + (size_t)lr * HD + 32 + half * 16), s0);
    s1 = wmma_bf16(aq0, *(const v16bf*)(kl + (size_t)(16 + lr) * HD + half * 16), s1);
    s1 = wmma_bf16(aq1, *(const v16bf*)(kl + (size_t)(16 + lr) * HD + 32 + half * 16), s1);

    float p0[8], p1[8], corr[8], tm[8], rs[8];
#pragma unroll
    for (int j = 0; j < 8; ++j) {
      s0[j] *= scale; s1[j] *= scale;
      tm[j] = fmaxf(s0[j], s1[j]);
    }
#pragma unroll
    for (int m = 1; m <= 8; m <<= 1) {
#pragma unroll
      for (int j = 0; j < 8; ++j) tm[j] = fmaxf(tm[j], __shfl_xor(tm[j], m, 32));
    }
#pragma unroll
    for (int j = 0; j < 8; ++j) {
      float mn = fmaxf(mrun[j], tm[j]);
      corr[j] = __expf(mrun[j] - mn);
      p0[j] = __expf(s0[j] - mn);
      p1[j] = __expf(s1[j] - mn);
      rs[j] = p0[j] + p1[j];
      mrun[j] = mn;
    }
#pragma unroll
    for (int m = 1; m <= 8; m <<= 1) {
#pragma unroll
      for (int j = 0; j < 8; ++j) rs[j] += __shfl_xor(rs[j], m, 32);
    }
#pragma unroll
    for (int j = 0; j < 8; ++j) {
      lrun[j] = lrun[j] * corr[j] + rs[j];
      float c = corr[j];
      o0[j] *= c; o1[j] *= c; o2[j] *= c; o3[j] *= c;
    }

    // ---- transpose P from D layout to A layout through LDS ----
#pragma unroll
    for (int j = 0; j < 8; ++j) {
      int row = j + 8 * half;            // D layout: m = vgpr + 8*(lane/16)
      pb[row * 32 + lr]       = (bf16_t)p0[j];
      pb[row * 32 + 16 + lr]  = (bf16_t)p1[j];
    }
    asm volatile("s_wait_dscnt 0" ::: "memory");  // wave-internal DS drain
    ABu up;
    up.h[0] = *(const v8bf*)(pb + lr * 32 + half * 16);
    up.h[1] = *(const v8bf*)(pb + lr * 32 + half * 16 + 8);
    v16bf pa = up.v;
    asm volatile("" ::: "memory");

    // ---- O += P * V  (Vt is [bh, d, n], contiguous in key index) ----
    const bf16_t* vb = Vt + ((size_t)bh * HD + lr) * NSEQ + (size_t)kt * 32 + half * 16;
    __builtin_prefetch(vb + 32, 0, 3);
    __builtin_prefetch(vb + 3 * 16 * NSEQ + 32, 0, 3);
    o0 = wmma_bf16(pa, *(const v16bf*)(vb + 0 * 16 * NSEQ), o0);
    o1 = wmma_bf16(pa, *(const v16bf*)(vb + 1 * 16 * NSEQ), o1);
    o2 = wmma_bf16(pa, *(const v16bf*)(vb + 2 * 16 * NSEQ), o2);
    o3 = wmma_bf16(pa, *(const v16bf*)(vb + 3 * 16 * NSEQ), o3);
  }

  // ---- normalize and write O back into [b, n, h*64+d] (bf16, feeds proj) ----
  int bb = bh / NHEAD, hh = bh % NHEAD;
#pragma unroll
  for (int j = 0; j < 8; ++j) {
    int m = qt * 16 + j + 8 * half;
    float invl = 1.0f / lrun[j];
    size_t rowoff = ((size_t)bb * NSEQ + m) * DIMC + hh * HD;
    O[rowoff + 0 * 16 + lr] = (bf16_t)(o0[j] * invl);
    O[rowoff + 1 * 16 + lr] = (bf16_t)(o1[j] * invl);
    O[rowoff + 2 * 16 + lr] = (bf16_t)(o2[j] * invl);
    O[rowoff + 3 * 16 + lr] = (bf16_t)(o3[j] * invl);
  }
}

// ---------------------------------------------------------------------------
// proj GEMM + bias + residual -> x1 (f32)
// ---------------------------------------------------------------------------
__global__ __launch_bounds__(256) void gemm_proj_kernel(const bf16_t* __restrict__ A,
                                                        const bf16_t* __restrict__ W,
                                                        const float* __restrict__ xres,
                                                        const float* __restrict__ bias,
                                                        float* __restrict__ x1) {
  const int NT = DIMC / 64;  // 12
  int wid = threadIdx.x >> 5, lane = threadIdx.x & 31;
  int half = lane >> 4, lr = lane & 15;
  long gw = (long)blockIdx.x * 8 + wid;
  int m0 = (int)(gw / NT) * 32;
  int n0 = (int)(gw % NT) * 64;
  v8f acc[8];
  zero_acc(acc);
  gemm_tile_32x64<DIMC>(A, W, m0, n0, half, lr, acc);
#pragma unroll
  for (int i = 0; i < 2; ++i) {
#pragma unroll
    for (int t = 0; t < 4; ++t) {
#pragma unroll
      for (int j = 0; j < 8; ++j) {
        int m = m0 + i * 16 + j + 8 * half;
        int col = n0 + t * 16 + lr;
        size_t idx = (size_t)m * DIMC + col;
        x1[idx] = xres[idx] + acc[i * 4 + t][j] + bias[col];
      }
    }
  }
}

// ---------------------------------------------------------------------------
// fc1 GEMM + bias + exact GELU -> bf16
// ---------------------------------------------------------------------------
__global__ __launch_bounds__(256) void gemm_fc1_kernel(const bf16_t* __restrict__ A,
                                                       const bf16_t* __restrict__ W,
                                                       const float* __restrict__ bias,
                                                       bf16_t* __restrict__ mid) {
  const int NT = HIDDEN / 64;  // 48
  int wid = threadIdx.x >> 5, lane = threadIdx.x & 31;
  int half = lane >> 4, lr = lane & 15;
  long gw = (long)blockIdx.x * 8 + wid;
  int m0 = (int)(gw / NT) * 32;
  int n0 = (int)(gw % NT) * 64;
  v8f acc[8];
  zero_acc(acc);
  gemm_tile_32x64<DIMC>(A, W, m0, n0, half, lr, acc);
#pragma unroll
  for (int i = 0; i < 2; ++i) {
#pragma unroll
    for (int t = 0; t < 4; ++t) {
#pragma unroll
      for (int j = 0; j < 8; ++j) {
        int m = m0 + i * 16 + j + 8 * half;
        int col = n0 + t * 16 + lr;
        float v = acc[i * 4 + t][j] + bias[col];
        float ge = 0.5f * v * (1.0f + erff(v * 0.70710678118654752f));
        mid[(size_t)m * HIDDEN + col] = (bf16_t)ge;
      }
    }
  }
}

// ---------------------------------------------------------------------------
// fc2 GEMM (K=3072) + bias + residual -> final f32 output
// ---------------------------------------------------------------------------
__global__ __launch_bounds__(256) void gemm_fc2_kernel(const bf16_t* __restrict__ A,
                                                       const bf16_t* __restrict__ W,
                                                       const float* __restrict__ bias,
                                                       const float* __restrict__ x1,
                                                       float* __restrict__ out) {
  const int NT = DIMC / 64;  // 12
  int wid = threadIdx.x >> 5, lane = threadIdx.x & 31;
  int half = lane >> 4, lr = lane & 15;
  long gw = (long)blockIdx.x * 8 + wid;
  int m0 = (int)(gw / NT) * 32;
  int n0 = (int)(gw % NT) * 64;
  v8f acc[8];
  zero_acc(acc);
  gemm_tile_32x64<HIDDEN>(A, W, m0, n0, half, lr, acc);
#pragma unroll
  for (int i = 0; i < 2; ++i) {
#pragma unroll
    for (int t = 0; t < 4; ++t) {
#pragma unroll
      for (int j = 0; j < 8; ++j) {
        int m = m0 + i * 16 + j + 8 * half;
        int col = n0 + t * 16 + lr;
        size_t idx = (size_t)m * DIMC + col;
        out[idx] = x1[idx] + acc[i * 4 + t][j] + bias[col];
      }
    }
  }
}

// ---------------------------------------------------------------------------
// Host-side orchestration
// ---------------------------------------------------------------------------
extern "C" void kernel_launch(void* const* d_in, const int* in_sizes, int n_in,
                              void* d_out, int out_size, void* d_ws, size_t ws_size,
                              hipStream_t stream) {
  (void)in_sizes; (void)n_in; (void)out_size; (void)ws_size;
  const float* x      = (const float*)d_in[0];
  const float* ln1_g  = (const float*)d_in[1];
  const float* ln1_b  = (const float*)d_in[2];
  const float* qkv_w  = (const float*)d_in[3];
  const float* proj_w = (const float*)d_in[4];
  const float* proj_b = (const float*)d_in[5];
  const float* ln2_g  = (const float*)d_in[6];
  const float* ln2_b  = (const float*)d_in[7];
  const float* fc1_w  = (const float*)d_in[8];
  const float* fc1_b  = (const float*)d_in[9];
  const float* fc2_w  = (const float*)d_in[10];
  const float* fc2_b  = (const float*)d_in[11];
  float* out = (float*)d_out;

  char* base = (char*)d_ws;
  size_t off = 0;
  auto alloc = [&](size_t bytes) -> void* {
    void* p = base + off;
    off = (off + bytes + 255) & ~(size_t)255;
    return p;
  };

  bf16_t* h1  = (bf16_t*)alloc((size_t)MROWS * DIMC * 2);        // LN1 out
  bf16_t* wq  = (bf16_t*)alloc((size_t)3 * DIMC * DIMC * 2);     // qkv_w bf16
  bf16_t* wp  = (bf16_t*)alloc((size_t)DIMC * DIMC * 2);         // proj_w bf16
  bf16_t* w1  = (bf16_t*)alloc((size_t)HIDDEN * DIMC * 2);       // fc1_w bf16
  bf16_t* w2  = (bf16_t*)alloc((size_t)DIMC * HIDDEN * 2);       // fc2_w bf16
  bf16_t* qb  = (bf16_t*)alloc((size_t)BHTOT * NSEQ * HD * 2);   // Q [bh,n,d]
  bf16_t* kb  = (bf16_t*)alloc((size_t)BHTOT * NSEQ * HD * 2);   // K [bh,n,d]
  bf16_t* vtb = (bf16_t*)alloc((size_t)BHTOT * HD * NSEQ * 2);   // V^T [bh,d,n]
  bf16_t* ob  = (bf16_t*)alloc((size_t)MROWS * DIMC * 2);        // attn out
  float*  x1  = (float*)alloc((size_t)MROWS * DIMC * 4);         // post-attn residual
  bf16_t* h2  = (bf16_t*)alloc((size_t)MROWS * DIMC * 2);        // LN2 out
  bf16_t* mid = (bf16_t*)alloc((size_t)MROWS * HIDDEN * 2);      // gelu(fc1)

  // weights -> bf16
  int nqkv = 3 * DIMC * DIMC, nproj = DIMC * DIMC, nfc = HIDDEN * DIMC;
  f32_to_bf16_kernel<<<(nqkv + 255) / 256, 256, 0, stream>>>(qkv_w, wq, nqkv);
  f32_to_bf16_kernel<<<(nproj + 255) / 256, 256, 0, stream>>>(proj_w, wp, nproj);
  f32_to_bf16_kernel<<<(nfc + 255) / 256, 256, 0, stream>>>(fc1_w, w1, nfc);
  f32_to_bf16_kernel<<<(nfc + 255) / 256, 256, 0, stream>>>(fc2_w, w2, nfc);

  // LN1
  ln_bf16_kernel<<<MROWS, 256, 0, stream>>>(x, ln1_g, ln1_b, h1);

  // QKV GEMM: (M/32)*(3C/64) waves, 8 per block
  gemm_qkv_kernel<<<(MROWS / 32) * ((3 * DIMC) / 64) / 8, 256, 0, stream>>>(h1, wq, qb, kb, vtb);

  // Flash attention: BHTOT * (NSEQ/16) waves / 8 per block
  attn_kernel<<<BHTOT * (NSEQ / 16) / 8, 256, 0, stream>>>(qb, kb, vtb, ob);

  // proj + residual
  gemm_proj_kernel<<<(MROWS / 32) * (DIMC / 64) / 8, 256, 0, stream>>>(ob, wp, x, proj_b, x1);

  // LN2
  ln_bf16_kernel<<<MROWS, 256, 0, stream>>>(x1, ln2_g, ln2_b, h2);

  // fc1 + GELU
  gemm_fc1_kernel<<<(MROWS / 32) * (HIDDEN / 64) / 8, 256, 0, stream>>>(h2, w1, fc1_b, mid);

  // fc2 + residual -> out
  gemm_fc2_kernel<<<(MROWS / 32) * (DIMC / 64) / 8, 256, 0, stream>>>(mid, w2, fc2_b, x1, out);
}
